// CTRNN_57354993271015
// MI455X (gfx1250) — matrix operand order repaired and checked by reading, not compiled, not run on hardware
//
#include <hip/hip_runtime.h>

typedef __bf16 bf16_t;
typedef __attribute__((ext_vector_type(16))) __bf16 v16bf;
typedef __attribute__((ext_vector_type(8)))  float  v8f;

#define ALPHA_C 0.2f
#define OMA_C   0.8f

static constexpr int Tn = 512;
static constexpr int Bn = 256;
static constexpr int In = 64;
static constexpr int Hn = 256;

// -------------------------------------------------------------------------
// Kernel 1: U[t,b,j] = ALPHA*(x[t,b,:] @ W_in[j,:]) + ALPHA*(b_in[j]+b_hh[j]+noise[t,j])
// Each wave computes a 16x64 slab (4 column tiles) reusing one A operand;
// K=64 -> 2 x v_wmma_f32_16x16x32_bf16 per tile. 16-row tiles never cross t.
// -------------------------------------------------------------------------
__global__ __launch_bounds__(256) void ctrnn_input_gemm(
    const float* __restrict__ x, const float* __restrict__ W_in,
    const float* __restrict__ b_in, const float* __restrict__ b_hh,
    const float* __restrict__ noise, float* __restrict__ U)
{
    const int lane = threadIdx.x & 31;
    const int wave = threadIdx.x >> 5;
    const int lh   = lane & 15;     // row (A) / col (B,D) within tile
    const int hi   = lane >> 4;     // lane-half selector

    const int grp     = blockIdx.x * 8 + wave;   // 32768 groups total
    const int rowTile = grp >> 2;                // 0..8191
    const int cg      = grp & 3;                 // column group (64 cols)
    const int rbase   = rowTile << 4;            // flattened (t,b) row base
    const int t       = rbase >> 8;              // 256 rows per t

    // A operand (shared by 4 col tiles): row M = lh;
    // K elems 0-7 -> kb+hi*8, elems 8-15 -> kb+16+hi*8
    const float* xrow = x + (size_t)(rbase + lh) * In;
    v16bf a0, a1;
    #pragma unroll
    for (int e = 0; e < 8; ++e) {
        a0[e]     = (bf16_t)xrow[hi * 8 + e];
        a0[8 + e] = (bf16_t)xrow[16 + hi * 8 + e];
        a1[e]     = (bf16_t)xrow[32 + hi * 8 + e];
        a1[8 + e] = (bf16_t)xrow[48 + hi * 8 + e];
    }

    #pragma unroll
    for (int s = 0; s < 4; ++s) {
        const int j = (cg << 6) + (s << 4) + lh;         // output column (N = lh)

        // B operand: B[k][j] = W_in[j][k]; K elems 0-7 -> kb+hi*8, elems 8-15 -> kb+16+hi*8
        const float* wrow = W_in + j * In;
        v16bf b0, b1;
        #pragma unroll
        for (int e = 0; e < 8; ++e) {
            b0[e]     = (bf16_t)wrow[hi * 8 + e];
            b0[8 + e] = (bf16_t)wrow[16 + hi * 8 + e];
            b1[e]     = (bf16_t)wrow[32 + hi * 8 + e];
            b1[8 + e] = (bf16_t)wrow[48 + hi * 8 + e];
        }

        v8f acc = {};
        acc = __builtin_amdgcn_wmma_f32_16x16x32_bf16(false, a0, false, b0,
                                                      (short)0, acc, false, false);
        acc = __builtin_amdgcn_wmma_f32_16x16x32_bf16(false, a1, false, b1,
                                                      (short)0, acc, false, false);

        const float bias = ALPHA_C * (b_in[j] + b_hh[j] + noise[t * Hn + j]);
        #pragma unroll
        for (int r = 0; r < 8; ++r) {
            const int row = rbase + hi * 8 + r;  // D: lane<16 -> M=r, lane>=16 -> M=8+r
            U[(size_t)row * Hn + j] = ALPHA_C * acc[r] + bias;
        }
    }
}

// -------------------------------------------------------------------------
// Kernel 2: sequential scan. 16 blocks (one per 16-row batch slice),
// 512 threads = 16 waves; wave w owns output column tile w (cols 16w..16w+15).
// W_hh^T B-operand lives in registers (8 x v16bf per wave, loaded once).
// h state lives in registers in the D layout; published per step as bf16 to a
// ping-pong LDS buffer (single barrier per step). U reads are read-once (NT),
// h writes stream past L2 (NT store) so the L2-resident U region stays hot;
// U[t+1] is prefetched under the matmul.
// -------------------------------------------------------------------------
__global__ __launch_bounds__(512) void ctrnn_scan(
    const float* __restrict__ hidden, const float* __restrict__ W_hh,
    float* __restrict__ out, float* __restrict__ h_last)
{
    __shared__ bf16_t hlds[2][16 * Hn];          // ping-pong, 16 KB total

    const int lane  = threadIdx.x & 31;
    const int wave  = threadIdx.x >> 5;          // 0..15 = owned column tile
    const int lh    = lane & 15;
    const int hi    = lane >> 4;
    const int bbase = blockIdx.x << 4;           // batch slice base
    const int j     = (wave << 4) + lh;          // owned column (N = lh)

    // --- load W_hh^T B-operand into registers, bf16, swizzled per ISA layout.
    // B[k][j] = W_hh[j][k]; lane holds K = kc*32 + hi*8 + (0..7) and kc*32 + 16 + hi*8 + (0..7) of column j.
    v16bf breg[8];
    #pragma unroll
    for (int kc = 0; kc < 8; ++kc) {
        const float* wr = W_hh + (size_t)j * Hn + (kc << 5) + (hi << 3);
        #pragma unroll
        for (int e = 0; e < 8; ++e) {
            breg[kc][e]     = (bf16_t)wr[e];
            breg[kc][8 + e] = (bf16_t)wr[16 + e];
        }
    }

    // --- init h state in D layout: VGPR r <-> batch row bbase + hi*8 + r, col j
    float hreg[8];
    #pragma unroll
    for (int r = 0; r < 8; ++r)
        hreg[r] = hidden[(size_t)(bbase + hi * 8 + r) * Hn + j];

    int p = 0;
    for (int t = 0; t < Tn; ++t) {
        // publish h as bf16, row-major [16][256]
        #pragma unroll
        for (int r = 0; r < 8; ++r)
            hlds[p][(hi * 8 + r) * Hn + j] = (bf16_t)hreg[r];
        __syncthreads();

        // U[t] for owned elements: read-once, issued before the matmul
        float* orow = out + (size_t)t * Bn * Hn;
        float u[8];
        #pragma unroll
        for (int r = 0; r < 8; ++r)
            u[r] = __builtin_nontemporal_load(
                &orow[(size_t)(bbase + hi * 8 + r) * Hn + j]);

        // warm up U[t+1] while the WMMAs run (global_prefetch_b8)
        if (t + 1 < Tn) {
            const float* nrow = out + (size_t)(t + 1) * Bn * Hn;
            #pragma unroll
            for (int r = 0; r < 8; ++r)
                __builtin_prefetch(&nrow[(size_t)(bbase + hi * 8 + r) * Hn + j], 0, 3);
        }

        // pre = h @ W_hh^T for owned 16x16 tile, K=256 ->
        // 8 WMMAs as two interleaved accumulation chains
        v8f acc0 = {}, acc1 = {};
        #pragma unroll
        for (int kc = 0; kc < 8; kc += 2) {
            const bf16_t* hr0 = &hlds[p][lh * Hn + (kc << 5)];
            v16bf aA;
            #pragma unroll
            for (int e = 0; e < 8; ++e) {
                aA[e]     = hr0[hi * 8 + e];
                aA[8 + e] = hr0[16 + hi * 8 + e];
            }
            acc0 = __builtin_amdgcn_wmma_f32_16x16x32_bf16(
                false, aA, false, breg[kc], (short)0, acc0, false, false);

            const bf16_t* hr1 = &hlds[p][lh * Hn + ((kc + 1) << 5)];
            v16bf aB;
            #pragma unroll
            for (int e = 0; e < 8; ++e) {
                aB[e]     = hr1[hi * 8 + e];
                aB[8 + e] = hr1[16 + hi * 8 + e];
            }
            acc1 = __builtin_amdgcn_wmma_f32_16x16x32_bf16(
                false, aB, false, breg[kc + 1], (short)0, acc1, false, false);
        }
        const v8f acc = acc0 + acc1;

        // epilogue: leaky-integrate + relu; stream h into out[t] (never re-read)
        #pragma unroll
        for (int r = 0; r < 8; ++r) {
            const size_t idx = (size_t)(bbase + hi * 8 + r) * Hn + j;
            float hn = fmaxf(OMA_C * hreg[r] + ALPHA_C * acc[r] + u[r], 0.0f);
            __builtin_nontemporal_store(hn, &orow[idx]);
            hreg[r] = hn;
        }
        p ^= 1;   // ping-pong: next write goes to the buffer last read 2 steps ago
    }

    #pragma unroll
    for (int r = 0; r < 8; ++r)
        h_last[(size_t)(bbase + hi * 8 + r) * Hn + j] = hreg[r];
}

extern "C" void kernel_launch(void* const* d_in, const int* in_sizes, int n_in,
                              void* d_out, int out_size, void* d_ws, size_t ws_size,
                              hipStream_t stream) {
    (void)in_sizes; (void)n_in; (void)out_size; (void)d_ws; (void)ws_size;
    const float* x      = (const float*)d_in[0];
    const float* hidden = (const float*)d_in[1];
    const float* noise  = (const float*)d_in[2];
    const float* W_in   = (const float*)d_in[3];
    const float* b_in   = (const float*)d_in[4];
    const float* W_hh   = (const float*)d_in[5];
    const float* b_hh   = (const float*)d_in[6];

    float* out    = (float*)d_out;                        // [T,B,H]
    float* h_last = out + (size_t)Tn * Bn * Hn;           // [B,H]

    // Phase 1: U -> out region (134 MB, memory-bound, fully parallel)
    ctrnn_input_gemm<<<4096, 256, 0, stream>>>(x, W_in, b_in, b_hh, noise, out);
    // Phase 2: sequential scan, batch-parallel across 16 workgroups
    ctrnn_scan<<<16, 512, 0, stream>>>(hidden, W_hh, out, h_last);
}
